// cross_att_12309376270837
// MI455X (gfx1250) — matrix operand order repaired and checked by reading, not compiled, not run on hardware
//
#include <hip/hip_runtime.h>
#include <math.h>

#define HNUM 8
#define DIMK 512
#define HH 65
#define WW 65
#define NPIX 4225          // 65*65
#define RWF 33             // rfft last-dim
#define DROW 2145          // 65*33
#define PP 4288            // pixel pad: multiple of 64
#define K9 4608            // 512*9 (im2col K)
#define DIMHID 4096

#define BM 128
#define BN 64
#define BK 32

typedef __attribute__((ext_vector_type(16))) _Float16 v16h;
typedef __attribute__((ext_vector_type(8)))  _Float16 v8h;
typedef __attribute__((ext_vector_type(8)))  float    v8f;

// ---------------------------------------------------------------------------
// WMMA GEMM with async-to-LDS double buffering.
// C[M][N] = A[M][K] * BT[N][K]^T (+ optional residual R). Operands f16,
// K-contiguous. Block = 256 threads (8 waves), block tile 128(M) x 64(N);
// wave w computes rows [m0+16w, m0+16w+16) x 64 cols (4 accumulators).
// Tiles staged to LDS with GLOBAL_LOAD_ASYNC_TO_LDS_B128 (ASYNCcnt),
// double-buffered: stage k+32 while computing k. All fragments are ds-loaded
// up front so the 4 WMMAs issue back-to-back behind one s_wait_dscnt.
// ---------------------------------------------------------------------------
__launch_bounds__(256)
__global__ void gemm_lds_async(const _Float16* __restrict__ A, int lda,
                               const _Float16* __restrict__ BT, int ldb,
                               float* __restrict__ C, int ldc,
                               const float* __restrict__ R, int Kdim) {
  __shared__ _Float16 sA[2][BM * BK];   // 2 x 8 KB
  __shared__ _Float16 sB[2][BN * BK];   // 2 x 4 KB

  const int tid  = threadIdx.x;
  const int lane = tid & 31;
  const int wave = tid >> 5;
  const int row  = lane & 15;
  const int grp  = lane >> 4;
  const int n0   = blockIdx.x * BN;
  const int m0   = blockIdx.y * BM;
  const int mw   = m0 + wave * 16;

  v8f acc[4];
#pragma unroll
  for (int t = 0; t < 4; ++t)
#pragma unroll
    for (int r = 0; r < 8; ++r)
      acc[t][r] = R ? R[(mw + r + grp * 8) * ldc + n0 + t * 16 + row] : 0.0f;

  // Stage one k-slab (A: 128x32, B: 64x32 halfs) into LDS buffer `buf`.
  auto stage = [&](int buf, int k0) {
    // A tile: 512 16-byte chunks, 2 per thread.
#pragma unroll
    for (int q = tid; q < (BM * BK) / 8; q += 256) {
      int r = q >> 2, c16 = q & 3;
      unsigned l = (unsigned)(size_t)&sA[buf][r * BK + c16 * 8];
      unsigned long long g =
          (unsigned long long)(A + (size_t)(m0 + r) * lda + k0 + c16 * 8);
      asm volatile("global_load_async_to_lds_b128 %0, %1, off"
                   :: "v"(l), "v"(g) : "memory");
    }
    // B tile: 256 16-byte chunks, 1 per thread.
    {
      int r = tid >> 2, c16 = tid & 3;
      unsigned l = (unsigned)(size_t)&sB[buf][r * BK + c16 * 8];
      unsigned long long g =
          (unsigned long long)(BT + (size_t)(n0 + r) * ldb + k0 + c16 * 8);
      asm volatile("global_load_async_to_lds_b128 %0, %1, off"
                   :: "v"(l), "v"(g) : "memory");
    }
  };

  stage(0, 0);
  int cur = 0;
  for (int k0 = 0; k0 < Kdim; k0 += BK) {
    asm volatile("s_wait_asynccnt 0" ::: "memory");  // own async writes landed
    __syncthreads();                                 // everyone's tiles ready
    if (k0 + BK < Kdim) stage(cur ^ 1, k0 + BK);     // prefetch next slab

    // Load A fragment + all 4 B fragments first (one ds clause, one wait),
    // then issue the 4 independent WMMAs back-to-back.
    union { v16h v; v8h h[2]; } a, b[4];
    const _Float16* pa = &sA[cur][(wave * 16 + row) * BK + grp * 8];
    a.h[0] = *(const v8h*)pa;
    a.h[1] = *(const v8h*)(pa + 16);
#pragma unroll
    for (int t = 0; t < 4; ++t) {
      const _Float16* pb = &sB[cur][(t * 16 + row) * BK + grp * 8];
      b[t].h[0] = *(const v8h*)pb;
      b[t].h[1] = *(const v8h*)(pb + 16);
    }
#pragma unroll
    for (int t = 0; t < 4; ++t)
      acc[t] = __builtin_amdgcn_wmma_f32_16x16x32_f16(
          false, a.v, false, b[t].v, (short)0, acc[t], false, false);
    cur ^= 1;
  }

#pragma unroll
  for (int t = 0; t < 4; ++t)
#pragma unroll
    for (int r = 0; r < 8; ++r)
      C[(mw + r + grp * 8) * ldc + n0 + t * 16 + row] = acc[t][r];
}

// ---------------------------------------------------------------------------
// K (512, 4225) -> KT f16 (PP, 512), LDS-tiled transpose, pad rows zeroed.
// ---------------------------------------------------------------------------
__global__ void transpose_cast_K(const float* __restrict__ K,
                                 _Float16* __restrict__ KT) {
  __shared__ float tile[32][33];
  int p0 = blockIdx.x * 32, c0 = blockIdx.y * 32;
  int tx = threadIdx.x, ty = threadIdx.y;
#pragma unroll
  for (int i = 0; i < 4; ++i) {
    int c = c0 + ty + i * 8, p = p0 + tx;
    tile[ty + i * 8][tx] = (p < NPIX) ? K[(size_t)c * NPIX + p] : 0.0f;
  }
  __syncthreads();
#pragma unroll
  for (int i = 0; i < 4; ++i) {
    int p = p0 + ty + i * 8, c = c0 + tx;
    if (p < PP) KT[(size_t)p * DIMK + c] = (_Float16)tile[tx][ty + i * 8];
  }
}

__global__ void cast_f16(const float* __restrict__ s, _Float16* __restrict__ d, int n) {
  int i = blockIdx.x * 256 + threadIdx.x;
  if (i < n) d[i] = (_Float16)s[i];
}

// q[j] = sum_c Q[c]*Wq[j,c] + bq[j]
__global__ void qvec_kernel(const float* __restrict__ Q, const float* __restrict__ Wq,
                            const float* __restrict__ bq, float* __restrict__ qv) {
  __shared__ float red[128];
  int j = blockIdx.x, t = threadIdx.x;
  float a = 0.f;
  for (int c = t; c < DIMK; c += 128) a += Q[c] * Wq[(size_t)j * DIMK + c];
  red[t] = a; __syncthreads();
  for (int s = 64; s > 0; s >>= 1) { if (t < s) red[t] += red[t + s]; __syncthreads(); }
  if (t == 0) qv[j] = red[0] + bq[j];
}

// w[h,c] = sum_d Wk[h*512+d, c] * q[h*512+d]   (collapsed K-projection)
__global__ void wvec_kernel(const float* __restrict__ Wk, const float* __restrict__ qv,
                            float* __restrict__ wv) {
  int i = blockIdx.x * 256 + threadIdx.x;
  if (i >= HNUM * DIMK) return;
  int h = i / DIMK, c = i % DIMK;
  float a = 0.f;
  for (int d = 0; d < DIMK; ++d)
    a += Wk[(size_t)(h * DIMK + d) * DIMK + c] * qv[h * DIMK + d];
  wv[i] = a;
}

__global__ void sbias_kernel(const float* __restrict__ bk, const float* __restrict__ qv,
                             float* __restrict__ sb) {
  __shared__ float red[64];
  int h = blockIdx.x, t = threadIdx.x;
  float a = 0.f;
  for (int d = t; d < DIMK; d += 64) a += bk[h * DIMK + d] * qv[h * DIMK + d];
  red[t] = a; __syncthreads();
  for (int s = 32; s > 0; s >>= 1) { if (t < s) red[t] += red[t + s]; __syncthreads(); }
  if (t == 0) sb[h] = red[0];
}

// S[h,p] = (sum_c K[c,p]*w[h,c] + sbias[h]) / sqrt(512)
__global__ void scores_kernel(const float* __restrict__ K, const float* __restrict__ wv,
                              const float* __restrict__ sb, float* __restrict__ S) {
  int i = blockIdx.x * 256 + threadIdx.x;
  if (i >= HNUM * PP) return;
  int h = i / PP, p = i % PP;
  float a = 0.f;
  if (p < NPIX) {
    for (int c = 0; c < DIMK; ++c) a += K[(size_t)c * NPIX + p] * wv[h * DIMK + c];
    a = (a + sb[h]) * 0.044194173824159216f;   // 1/sqrt(512)
  }
  S[i] = a;
}

// im2col with fused relu: G[p][cin*9 + ky*3+kx] = relu(F1[cin][pixel shifted]), pad=1
__global__ void im2col_kernel(const float* __restrict__ F1, _Float16* __restrict__ G) {
  int i = blockIdx.x * 256 + threadIdx.x;
  if (i >= PP * DIMK) return;
  int p = i / DIMK, cin = i % DIMK;
  _Float16* g = G + (size_t)p * K9 + cin * 9;
  if (p >= NPIX) {
#pragma unroll
    for (int t = 0; t < 9; ++t) g[t] = (_Float16)0.f;
    return;
  }
  int y = p / WW, x = p % WW;
#pragma unroll
  for (int ky = 0; ky < 3; ++ky)
#pragma unroll
    for (int kx = 0; kx < 3; ++kx) {
      int yy = y + ky - 1, xx = x + kx - 1;
      float v = 0.f;
      if (yy >= 0 && yy < HH && xx >= 0 && xx < WW) {
        v = F1[(size_t)cin * PP + yy * WW + xx];
        v = v > 0.f ? v : 0.f;
      }
      g[ky * 3 + kx] = (_Float16)v;
    }
}

// F3[h,p] = relu(sum_c w3[h,c] * relu(F2[c,p]))
__global__ void headconv_kernel(const float* __restrict__ F2, const float* __restrict__ w3,
                                float* __restrict__ F3) {
  int i = blockIdx.x * 256 + threadIdx.x;
  if (i >= HNUM * PP) return;
  int h = i / PP, p = i % PP;
  float a = 0.f;
  for (int c = 0; c < DIMK; ++c) {
    float v = F2[(size_t)c * PP + p];
    v = v > 0.f ? v : 0.f;
    a += w3[h * DIMK + c] * v;
  }
  F3[i] = a > 0.f ? a : 0.f;
}

// D[h,j] = sum_p F3[h,p] * Wd[j,p] + bd[j]   (block per j, 8 head accumulators)
__global__ void downproj_kernel(const float* __restrict__ F3, const float* __restrict__ Wd,
                                const float* __restrict__ bd, float* __restrict__ D) {
  __shared__ float red[8][256];
  int j = blockIdx.x, t = threadIdx.x;
  float acc[8] = {0, 0, 0, 0, 0, 0, 0, 0};
  for (int p = t; p < NPIX; p += 256) {
    float w = Wd[(size_t)j * NPIX + p];
#pragma unroll
    for (int h = 0; h < 8; ++h) acc[h] += F3[h * PP + p] * w;
  }
#pragma unroll
  for (int h = 0; h < 8; ++h) red[h][t] = acc[h];
  __syncthreads();
  for (int s = 128; s > 0; s >>= 1) {
    if (t < s)
#pragma unroll
      for (int h = 0; h < 8; ++h) red[h][t] += red[h][t + s];
    __syncthreads();
  }
  if (t < 8) D[t * DROW + j] = red[t][0] + bd[j];
}

// ---- separable naive DFTs on 65 (65 = 5*13; N^2 cost is negligible here) ----
__global__ void dft_rows_kernel(const float* __restrict__ S, float* __restrict__ B1) {
  int i = blockIdx.x * 256 + threadIdx.x;
  if (i >= HNUM * HH * RWF) return;
  int v = i % RWF, y = (i / RWF) % HH, h = i / (RWF * HH);
  const float* a = S + h * PP + y * WW;
  float re = 0.f, im = 0.f, w = -6.283185307179586f * v / 65.0f;
  for (int x = 0; x < WW; ++x) {
    float s, c; __sincosf(w * x, &s, &c);
    re += a[x] * c; im += a[x] * s;
  }
  B1[2 * i] = re; B1[2 * i + 1] = im;
}

__global__ void dft_cols_kernel(const float* __restrict__ B1, float* __restrict__ Af) {
  int i = blockIdx.x * 256 + threadIdx.x;
  if (i >= HNUM * HH * RWF) return;
  int v = i % RWF, u = (i / RWF) % HH, h = i / (RWF * HH);
  float re = 0.f, im = 0.f, w = -6.283185307179586f * u / 65.0f;
  for (int y = 0; y < HH; ++y) {
    float s, c; __sincosf(w * y, &s, &c);
    int b = 2 * ((h * HH + y) * RWF + v);
    float br = B1[b], bi = B1[b + 1];
    re += br * c - bi * s;
    im += br * s + bi * c;
  }
  Af[2 * i] = re; Af[2 * i + 1] = im;
}

__global__ void specmod_kernel(float* __restrict__ Af, const float* __restrict__ Da,
                               const float* __restrict__ Dp) {
  int i = blockIdx.x * 256 + threadIdx.x;
  if (i >= HNUM * HH * RWF) return;
  float re = Af[2 * i], im = Af[2 * i + 1];
  float mag = sqrtf(re * re + im * im);
  float ang = atan2f(im, re);
  int v = i % RWF, u = (i / RWF) % HH, h = i / (RWF * HH);
  float am = Da[h * DROW + u * RWF + v] * mag;
  float ph = Dp[h * DROW + u * RWF + v] * ang;
  Af[2 * i] = am * cosf(ph);
  Af[2 * i + 1] = am * sinf(ph);
}

__global__ void idft_cols_kernel(const float* __restrict__ Af, float* __restrict__ C1) {
  int i = blockIdx.x * 256 + threadIdx.x;
  if (i >= HNUM * HH * RWF) return;
  int v = i % RWF, y = (i / RWF) % HH, h = i / (RWF * HH);
  float re = 0.f, im = 0.f, w = 6.283185307179586f * y / 65.0f;
  for (int u = 0; u < HH; ++u) {
    float s, c; __sincosf(w * u, &s, &c);
    int b = 2 * ((h * HH + u) * RWF + v);
    float ar = Af[b], ai = Af[b + 1];
    re += ar * c - ai * s;
    im += ar * s + ai * c;
  }
  C1[2 * i] = re; C1[2 * i + 1] = im;
}

// Hermitian reconstruction along x + 1/(65*65) backward-norm scaling
__global__ void idft_rows_kernel(const float* __restrict__ C1, float* __restrict__ S) {
  int i = blockIdx.x * 256 + threadIdx.x;
  if (i >= HNUM * NPIX) return;
  int x = i % WW, y = (i / WW) % HH, h = i / (WW * HH);
  const float* c1 = C1 + 2 * (size_t)((h * HH + y) * RWF);
  float acc = c1[0];
  float w = 6.283185307179586f * x / 65.0f;
  for (int v = 1; v < RWF; ++v) {
    float s, c; __sincosf(w * v, &s, &c);
    acc += 2.f * (c1[2 * v] * c - c1[2 * v + 1] * s);
  }
  S[h * PP + y * WW + x] = acc * (1.0f / 4225.0f);
}

__global__ void softmax_kernel(float* __restrict__ S) {
  __shared__ float red[256];
  int h = blockIdx.x, t = threadIdx.x;
  float m = -1e30f;
  for (int p = t; p < NPIX; p += 256) m = fmaxf(m, S[h * PP + p]);
  red[t] = m; __syncthreads();
  for (int s = 128; s > 0; s >>= 1) { if (t < s) red[t] = fmaxf(red[t], red[t + s]); __syncthreads(); }
  m = red[0]; __syncthreads();
  float sum = 0.f;
  for (int p = t; p < NPIX; p += 256) sum += __expf(S[h * PP + p] - m);
  red[t] = sum; __syncthreads();
  for (int s = 128; s > 0; s >>= 1) { if (t < s) red[t] += red[t + s]; __syncthreads(); }
  float inv = 1.0f / red[0];
  for (int p = t; p < NPIX; p += 256) S[h * PP + p] = __expf(S[h * PP + p] - m) * inv;
}

// t[h,c] = sum_p S[h,p] * K[c,p]   (collapsed V-projection, part 1)
__global__ void attnv_kernel(const float* __restrict__ S, const float* __restrict__ K,
                             float* __restrict__ tvec) {
  __shared__ float red[8][256];
  int c = blockIdx.x, t = threadIdx.x;
  float acc[8] = {0, 0, 0, 0, 0, 0, 0, 0};
  for (int p = t; p < NPIX; p += 256) {
    float kv = K[(size_t)c * NPIX + p];
#pragma unroll
    for (int h = 0; h < 8; ++h) acc[h] += S[h * PP + p] * kv;
  }
#pragma unroll
  for (int h = 0; h < 8; ++h) red[h][t] = acc[h];
  __syncthreads();
  for (int s = 128; s > 0; s >>= 1) {
    if (t < s)
#pragma unroll
      for (int h = 0; h < 8; ++h) red[h][t] += red[h][t + s];
    __syncthreads();
  }
  if (t < 8) tvec[t * DIMK + c] = red[t][0];
}

// x[j] = sum_c t[h(j),c] * Wv[j,c] + bv[j]   (softmax sums to 1 -> bv additive)
__global__ void xout_kernel(const float* __restrict__ tvec, const float* __restrict__ Wv,
                            const float* __restrict__ bv, float* __restrict__ x) {
  __shared__ float red[128];
  int j = blockIdx.x, t = threadIdx.x, h = j / DIMK;
  float a = 0.f;
  for (int c = t; c < DIMK; c += 128) a += tvec[h * DIMK + c] * Wv[(size_t)j * DIMK + c];
  red[t] = a; __syncthreads();
  for (int s = 64; s > 0; s >>= 1) { if (t < s) red[t] += red[t + s]; __syncthreads(); }
  if (t == 0) x[j] = red[0] + bv[j];
}

__global__ void final_kernel(const float* __restrict__ x, const float* __restrict__ Wo,
                             const float* __restrict__ bo, float* __restrict__ out) {
  __shared__ float red[256];
  int o = blockIdx.x, t = threadIdx.x;
  float a = 0.f;
  for (int j = t; j < DIMHID; j += 256) a += x[j] * Wo[(size_t)o * DIMHID + j];
  red[t] = a; __syncthreads();
  for (int s = 128; s > 0; s >>= 1) { if (t < s) red[t] += red[t + s]; __syncthreads(); }
  if (t == 0) out[o] = red[0] + bo[o];
}

// ---------------------------------------------------------------------------
extern "C" void kernel_launch(void* const* d_in, const int* in_sizes, int n_in,
                              void* d_out, int out_size, void* d_ws, size_t ws_size,
                              hipStream_t stream) {
  const float* Q    = (const float*)d_in[0];
  const float* Kt   = (const float*)d_in[1];
  const float* Wq   = (const float*)d_in[2];
  const float* bq   = (const float*)d_in[3];
  const float* Wk   = (const float*)d_in[4];
  const float* bk   = (const float*)d_in[5];
  const float* Wv   = (const float*)d_in[6];
  const float* bv   = (const float*)d_in[7];
  const float* Wo   = (const float*)d_in[8];
  const float* bo   = (const float*)d_in[9];
  const float* amp1 = (const float*)d_in[10];
  const float* amp2 = (const float*)d_in[11];
  const float* amp3 = (const float*)d_in[12];
  const float* pha1 = (const float*)d_in[13];
  const float* pha2 = (const float*)d_in[14];
  const float* pha3 = (const float*)d_in[15];
  const float* Wd1  = (const float*)d_in[16];
  const float* bd1  = (const float*)d_in[17];
  const float* Wd2  = (const float*)d_in[18];
  const float* bd2  = (const float*)d_in[19];
  float* out = (float*)d_out;

  char* w = (char*)d_ws;
  auto alloc = [&](size_t bytes) -> void* {
    void* p = (void*)w;
    w += (bytes + 255) & ~(size_t)255;
    return p;
  };
  _Float16* KT  = (_Float16*)alloc((size_t)PP * DIMK * 2);
  _Float16* A1h = (_Float16*)alloc((size_t)DIMK * DIMK * 2);
  _Float16* A2h = (_Float16*)alloc((size_t)DIMK * K9 * 2);
  float*    F1  = (float*)alloc((size_t)DIMK * PP * 4);
  float*    F2  = (float*)alloc((size_t)DIMK * PP * 4);
  _Float16* G   = (_Float16*)alloc((size_t)PP * K9 * 2);
  float*    F3  = (float*)alloc((size_t)HNUM * PP * 4);
  float*    Da  = (float*)alloc((size_t)HNUM * DROW * 4);
  float*    Dp  = (float*)alloc((size_t)HNUM * DROW * 4);
  float*    qv  = (float*)alloc(DIMHID * 4);
  float*    wv  = (float*)alloc(HNUM * DIMK * 4);
  float*    sb  = (float*)alloc(HNUM * 4);
  float*    S   = (float*)alloc((size_t)HNUM * PP * 4);
  float*    B1  = (float*)alloc((size_t)HNUM * HH * RWF * 2 * 4);
  float*    Af  = (float*)alloc((size_t)HNUM * HH * RWF * 2 * 4);
  float*    C1  = (float*)alloc((size_t)HNUM * HH * RWF * 2 * 4);
  float*    tv  = (float*)alloc(HNUM * DIMK * 4);
  float*    xv  = (float*)alloc(DIMHID * 4);

  const int NFREQ = HNUM * HH * RWF;

  // Stage K as f16 transposed (BT operand of 1x1-conv GEMM)
  transpose_cast_K<<<dim3(PP / 32, DIMK / 32), dim3(32, 8), 0, stream>>>(Kt, KT);

  // Collapsed attention-score path
  qvec_kernel<<<DIMHID, 128, 0, stream>>>(Q, Wq, bq, qv);
  wvec_kernel<<<(HNUM * DIMK + 255) / 256, 256, 0, stream>>>(Wk, qv, wv);
  sbias_kernel<<<HNUM, 64, 0, stream>>>(bk, qv, sb);
  scores_kernel<<<(HNUM * PP + 255) / 256, 256, 0, stream>>>(Kt, wv, sb, S);

  // Filter branch (shared buffers, run twice sequentially)
  auto run_branch = [&](const float* w1, const float* w2, const float* w3,
                        const float* Wd, const float* bd, float* D) {
    cast_f16<<<(DIMK * DIMK + 255) / 256, 256, 0, stream>>>(w1, A1h, DIMK * DIMK);
    cast_f16<<<(DIMK * K9 + 255) / 256, 256, 0, stream>>>(w2, A2h, DIMK * K9);
    // 1x1 conv: F1(512,PP) = A1(512,512) * KT(PP,512)^T
    gemm_lds_async<<<dim3(PP / BN, DIMK / BM), 256, 0, stream>>>(
        A1h, DIMK, KT, DIMK, F1, PP, nullptr, DIMK);
    im2col_kernel<<<(PP * DIMK + 255) / 256, 256, 0, stream>>>(F1, G);
    // 3x3 conv + residual: F2 = A2(512,4608) * G(PP,4608)^T + F1
    gemm_lds_async<<<dim3(PP / BN, DIMK / BM), 256, 0, stream>>>(
        A2h, K9, G, K9, F2, PP, F1, K9);
    headconv_kernel<<<(HNUM * PP + 255) / 256, 256, 0, stream>>>(F2, w3, F3);
    downproj_kernel<<<DROW, 256, 0, stream>>>(F3, Wd, bd, D);
  };
  run_branch(amp1, amp2, amp3, Wd1, bd1, Da);
  run_branch(pha1, pha2, pha3, Wd2, bd2, Dp);

  // Spectral modulation: rfft2 -> amp/phase remix -> irfft2 (separable DFTs)
  dft_rows_kernel<<<(NFREQ + 255) / 256, 256, 0, stream>>>(S, B1);
  dft_cols_kernel<<<(NFREQ + 255) / 256, 256, 0, stream>>>(B1, Af);
  specmod_kernel<<<(NFREQ + 255) / 256, 256, 0, stream>>>(Af, Da, Dp);
  idft_cols_kernel<<<(NFREQ + 255) / 256, 256, 0, stream>>>(Af, C1);
  idft_rows_kernel<<<(HNUM * NPIX + 255) / 256, 256, 0, stream>>>(C1, S);

  // Softmax + collapsed V path + output projection
  softmax_kernel<<<HNUM, 256, 0, stream>>>(S);
  attnv_kernel<<<DIMK, 256, 0, stream>>>(S, Kt, tv);
  xout_kernel<<<DIMHID, 128, 0, stream>>>(tv, Wv, bv, xv);
  final_kernel<<<DIMK, 256, 0, stream>>>(xv, Wo, bo, out);
}
